// SwinTransformerBlock_55619826483749
// MI455X (gfx1250) — matrix operand order repaired and checked by reading, not compiled, not run on hardware
//
#include <hip/hip_runtime.h>
#include <hip/hip_bf16.h>
#include <cstdint>

// ---------------------------------------------------------------------------
// Swin block for MI455X (gfx1250, wave32, WMMA 16x16x32 f16 -> f32 acc)
// All fragment traffic via contiguous 128-bit loads; weights repacked into
// WMMA B-fragment order so every B fragment is one aligned 32-byte read.
// ---------------------------------------------------------------------------
#define B_  32
#define H_  56
#define W_  56
#define C_  256
#define NH_ 8
#define HD_ 32           // head dim
#define WS_ 7
#define SS_ 3
#define NTOK_ (B_*H_*W_) // 100352, divisible by 256
#define NWIN_ (B_*64)    // 2048 windows
#define HID_  (4*C_)     // 1024

typedef __attribute__((ext_vector_type(16))) _Float16 v16h;
typedef __attribute__((ext_vector_type(8)))  _Float16 v8h;
typedef __attribute__((ext_vector_type(8)))  float    v8f;

__device__ __forceinline__ v8f wmma_f16(v16h a, v16h b, v8f c) {
  return __builtin_amdgcn_wmma_f32_16x16x32_f16(false, a, false, b, (short)0, c,
                                                false, false);
}

// ---------------- WMMA fragment loaders (contiguous k = 16*hf + e, A and B alike) --------
// A: 16x32 f16 tile, row-major with leading dim lda (multiple of 8).
// Per lane: two contiguous 16-byte chunks -> b128 loads.
__device__ __forceinline__ v16h load_a16(const _Float16* __restrict__ A, int lda) {
  int lane = threadIdx.x & 31;
  int m = lane & 15, hf = lane >> 4;
  const _Float16* p = A + (size_t)m * lda + hf * 16;
  v8h lo = *(const v8h*)p;         // K = hf*16 + 0..7
  v8h hi = *(const v8h*)(p + 8);   // K = hf*16 + 8..15
  v16h a;
#pragma unroll
  for (int e = 0; e < 8; ++e) { a[e] = lo[e]; a[8 + e] = hi[e]; }
  return a;
}

// B fragment from pre-packed weights: tile = 512 halves in lane-major order,
// one aligned 32-byte read per lane.
__device__ __forceinline__ v16h load_b_packed(const _Float16* __restrict__ Wp) {
  int lane = threadIdx.x & 31;
  return *(const v16h*)(Wp + (size_t)lane * 16);
}

// Per-wave 32(M) x 64(N) tile: two A strips share four B fragments.
__device__ __forceinline__ void wmma_block(const _Float16* __restrict__ A0,
                                           const _Float16* __restrict__ A1,
                                           const _Float16* __restrict__ Wp,
                                           int K, int nt0, v8f acc[2][4]) {
  int ktiles = K >> 5;
  for (int kt = 0; kt < ktiles; ++kt) {
    if (kt + 1 < ktiles) {
      __builtin_prefetch(A0 + (kt + 1) * 32, 0, 1);   // global_prefetch_b8
      __builtin_prefetch(A1 + (kt + 1) * 32, 0, 1);
    }
    v16h a0 = load_a16(A0 + kt * 32, K);
    v16h a1 = load_a16(A1 + kt * 32, K);
#pragma unroll
    for (int t = 0; t < 4; ++t) {
      v16h b = load_b_packed(Wp + ((size_t)(nt0 + t) * ktiles + kt) * 512);
      acc[0][t] = wmma_f16(a0, b, acc[0][t]);
      acc[1][t] = wmma_f16(a1, b, acc[1][t]);
    }
  }
}

// ---------------- weight cast + fragment repack ----------------------------
// Packs f32 weight [K x Nc] into f16 WMMA-B fragment tiles:
// tile (nt, kt) at ((nt*ktiles + kt)*512), lane-major 16 halves each.
__global__ void pack_w16(const float* __restrict__ w, _Float16* __restrict__ out,
                         int K, int Nc) {
  int t = blockIdx.x * 256 + threadIdx.x;    // one lane-slot per thread
  int total = (K >> 5) * (Nc >> 4) * 32;
  if (t >= total) return;
  int lane = t & 31;
  int tile = t >> 5;
  int ktiles = K >> 5;
  int kt = tile % ktiles;
  int nt = tile / ktiles;
  int n = lane & 15, hf = lane >> 4;
  v16h vv;
#pragma unroll
  for (int j = 0; j < 8; ++j) {
    int k = kt * 32 + hf * 16 + 2 * j;
    vv[2 * j]     = (_Float16)w[(size_t)k * Nc + nt * 16 + n];
    vv[2 * j + 1] = (_Float16)w[(size_t)(k + 1) * Nc + nt * 16 + n];
  }
  *(v16h*)(out + (size_t)t * 16) = vv;
}

// ---------------- layernorm ------------------------------------------------
// SHIFT=true: block = window-token row, gather via roll(-3)+partition map.
template <bool SHIFT>
__global__ void lnorm_kernel(const float* __restrict__ in, const float* __restrict__ g,
                             const float* __restrict__ bta, _Float16* __restrict__ out) {
  int row = blockIdx.x;
  int c = threadIdx.x;
  size_t src;
  if (SHIFT) {
    int win = row / 49, n = row % 49;
    int bb = win >> 6, wi = win & 63;
    int wh = wi >> 3, ww = wi & 7;
    int h = (wh * WS_ + n / WS_ + SS_) % H_;
    int w = (ww * WS_ + n % WS_ + SS_) % W_;
    src = (((size_t)bb * H_ + h) * W_ + w) * C_;
  } else {
    src = (size_t)row * C_;
  }
  float v = in[src + c];
  __shared__ float red[256];
  red[c] = v; __syncthreads();
  for (int s = 128; s > 0; s >>= 1) { if (c < s) red[c] += red[c + s]; __syncthreads(); }
  float mean = red[0] * (1.0f / 256.0f);
  __syncthreads();
  float d = v - mean;
  red[c] = d * d; __syncthreads();
  for (int s = 128; s > 0; s >>= 1) { if (c < s) red[c] += red[c + s]; __syncthreads(); }
  float rstd = rsqrtf(red[0] * (1.0f / 256.0f) + 1e-5f);
  out[(size_t)row * C_ + c] = (_Float16)(d * rstd * g[c] + bta[c]);
}

// ---------------- GEMM kernels (block tile 256M x 64N, 8 waves) ------------
template <bool GELU>
__global__ void gemm_bias_f16(const _Float16* __restrict__ A, const _Float16* __restrict__ Wp,
                              const float* __restrict__ bias, _Float16* __restrict__ out,
                              int Nc, int K) {
  int wave = threadIdx.x >> 5;
  int mBase = blockIdx.y * 256 + wave * 16;   // second strip at +128
  int nt0 = blockIdx.x * 4;
  v8f acc[2][4] = {};
  wmma_block(A + (size_t)mBase * K, A + (size_t)(mBase + 128) * K, Wp, K, nt0, acc);
  int lane = threadIdx.x & 31;
  int n = lane & 15, hf = lane >> 4;
#pragma unroll
  for (int s = 0; s < 2; ++s)
#pragma unroll
    for (int t = 0; t < 4; ++t) {
      int col = (nt0 + t) * 16 + n;
      float bsv = bias[col];
#pragma unroll
      for (int r = 0; r < 8; ++r) {
        int row = mBase + s * 128 + hf * 8 + r;
        float v = acc[s][t][r] + bsv;
        if (GELU) v = 0.5f * v * (1.0f + erff(v * 0.70710678118654752f));
        out[(size_t)row * Nc + col] = (_Float16)v;
      }
    }
}

// proj GEMM: rows in window-token order; epilogue scatters through
// window-reverse + roll(+3) and adds residual x -> x2 (fp32).
__global__ void gemm_proj_scatter(const _Float16* __restrict__ A, const _Float16* __restrict__ Wp,
                                  const float* __restrict__ bias, const float* __restrict__ x,
                                  float* __restrict__ x2, int Nc, int K) {
  int wave = threadIdx.x >> 5;
  int mBase = blockIdx.y * 256 + wave * 16;
  int nt0 = blockIdx.x * 4;
  v8f acc[2][4] = {};
  wmma_block(A + (size_t)mBase * K, A + (size_t)(mBase + 128) * K, Wp, K, nt0, acc);
  int lane = threadIdx.x & 31;
  int n = lane & 15, hf = lane >> 4;
#pragma unroll
  for (int s = 0; s < 2; ++s)
#pragma unroll
    for (int t = 0; t < 4; ++t) {
      int col = (nt0 + t) * 16 + n;
      float bsv = bias[col];
#pragma unroll
      for (int r = 0; r < 8; ++r) {
        int row = mBase + s * 128 + hf * 8 + r;
        int win = row / 49, nTok = row % 49;
        int bb = win >> 6, wi = win & 63;
        int wh = wi >> 3, ww = wi & 7;
        int h = (wh * WS_ + nTok / WS_ + SS_) % H_;
        int w = (ww * WS_ + nTok % WS_ + SS_) % W_;
        size_t idx = (((size_t)bb * H_ + h) * W_ + w) * C_ + col;
        x2[idx] = x[idx] + acc[s][t][r] + bsv;
      }
    }
}

// MLP2 GEMM: out = A @ W + bias + x2 (residual), fp32 result.
__global__ void gemm_mlp2_out(const _Float16* __restrict__ A, const _Float16* __restrict__ Wp,
                              const float* __restrict__ bias, const float* __restrict__ x2,
                              float* __restrict__ out, int Nc, int K) {
  int wave = threadIdx.x >> 5;
  int mBase = blockIdx.y * 256 + wave * 16;
  int nt0 = blockIdx.x * 4;
  v8f acc[2][4] = {};
  wmma_block(A + (size_t)mBase * K, A + (size_t)(mBase + 128) * K, Wp, K, nt0, acc);
  int lane = threadIdx.x & 31;
  int n = lane & 15, hf = lane >> 4;
#pragma unroll
  for (int s = 0; s < 2; ++s)
#pragma unroll
    for (int t = 0; t < 4; ++t) {
      int col = (nt0 + t) * 16 + n;
      float bsv = bias[col];
#pragma unroll
      for (int r = 0; r < 8; ++r) {
        int row = mBase + s * 128 + hf * 8 + r;
        size_t idx = (size_t)row * Nc + col;
        out[idx] = acc[s][t][r] + bsv + x2[idx];
      }
    }
}

// ---------------- attention kernel -----------------------------------------
// One block per (window, head). 128 threads = 4 wave32s. All LDS fragment
// reads are contiguous 32-byte runs (V is staged transposed).
__global__ void attn_kernel(const _Float16* __restrict__ qkv,
                            const float* __restrict__ bias_table,
                            _Float16* __restrict__ attn_out) {
  int win = blockIdx.x;
  int head = blockIdx.y;
  int tid = threadIdx.x;
  int wi = win & 63;
  int wh = wi >> 3, ww = wi & 7;

  __shared__ __align__(32) _Float16 q_s[64 * 32];   // [tok][dim]
  __shared__ __align__(32) _Float16 k_s[64 * 32];   // [tok][dim]
  __shared__ __align__(32) _Float16 vt_s[32 * 64];  // [dim][tok] (transposed)
  __shared__ __align__(32) float    s_s[64 * 64];
  __shared__ __align__(32) _Float16 p_s[64 * 64];

  const float scale = 0.17677669529663687f;  // 1/sqrt(32)
  for (int idx = tid; idx < 64 * 32; idx += 128) {
    int n = idx >> 5, d = idx & 31;
    _Float16 qv = (_Float16)0.f, kv = (_Float16)0.f, vv = (_Float16)0.f;
    if (n < 49) {
      size_t base = ((size_t)win * 49 + n) * (3 * C_) + head * HD_ + d;
      qv = (_Float16)((float)qkv[base] * scale);
      kv = qkv[base + C_];
      vv = qkv[base + 2 * C_];
    }
    q_s[idx] = qv;
    k_s[idx] = kv;
    vt_s[d * 64 + n] = vv;
  }
  __syncthreads();

  int wave = tid >> 5;
  int lane = tid & 31;
  int nl = lane & 15, hf = lane >> 4;

  // S = Q K^T : wave owns a 16-row strip, 4 column tiles, K=32.
  {
    int mBase = wave * 16;
    v16h a = load_a16(q_s + mBase * 32, 32);
#pragma unroll
    for (int t = 0; t < 4; ++t) {
      // B[kk][n] = K[t*16+n][kk]: contiguous 16 halves per lane.
      v16h b = *(const v16h*)(k_s + (t * 16 + nl) * 32 + hf * 16);
      v8f acc = {};
      acc = wmma_f16(a, b, acc);
#pragma unroll
      for (int r = 0; r < 8; ++r)
        s_s[(mBase + hf * 8 + r) * 64 + t * 16 + nl] = acc[r];
    }
  }
  __syncthreads();

  // softmax rows: thread i (<64) owns row i; rel-pos bias + shift mask inline.
  if (tid < 64) {
    int i = tid;
    if (i < 49) {
      int ri = i / WS_, ci = i % WS_;
      int hI = wh * WS_ + ri, wI = ww * WS_ + ci;
      int li = ((hI < 49) ? 0 : (hI < 53) ? 1 : 2) * 3 +
               ((wI < 49) ? 0 : (wI < 53) ? 1 : 2);
      float rowv[49];
      float mx = -1e30f;
      for (int j = 0; j < 49; ++j) {
        int rj = j / WS_, cj = j % WS_;
        int hJ = wh * WS_ + rj, wJ = ww * WS_ + cj;
        int lj = ((hJ < 49) ? 0 : (hJ < 53) ? 1 : 2) * 3 +
                 ((wJ < 49) ? 0 : (wJ < 53) ? 1 : 2);
        int rel = (ri - rj + WS_ - 1) * (2 * WS_ - 1) + (ci - cj + WS_ - 1);
        float v = s_s[i * 64 + j] + bias_table[rel * NH_ + head] +
                  ((li != lj) ? -100.0f : 0.0f);
        rowv[j] = v;
        mx = fmaxf(mx, v);
      }
      float sum = 0.f;
      for (int j = 0; j < 49; ++j) { float e = __expf(rowv[j] - mx); rowv[j] = e; sum += e; }
      float inv = 1.0f / sum;
      for (int j = 0; j < 49; ++j) p_s[i * 64 + j] = (_Float16)(rowv[j] * inv);
      for (int j = 49; j < 64; ++j) p_s[i * 64 + j] = (_Float16)0.f;
    } else {
      for (int j = 0; j < 64; ++j) p_s[i * 64 + j] = (_Float16)0.f;
    }
  }
  __syncthreads();

  // O = P @ V : 16-row strip per wave, 2 column tiles (hd=32), K=64.
  {
    int mBase = wave * 16;
    v8f acc[2] = {};
    for (int k0 = 0; k0 < 64; k0 += 32) {
      v16h a = load_a16(p_s + mBase * 64 + k0, 64);
#pragma unroll
      for (int t = 0; t < 2; ++t) {
        // B[k][n] = V[k][n] = vt[n][k]: contiguous 16 halves per lane.
        v16h b = *(const v16h*)(vt_s + (t * 16 + nl) * 64 + k0 + hf * 16);
        acc[t] = wmma_f16(a, b, acc[t]);
      }
    }
#pragma unroll
    for (int t = 0; t < 2; ++t)
#pragma unroll
      for (int r = 0; r < 8; ++r) {
        int i = mBase + hf * 8 + r;
        if (i < 49)
          attn_out[((size_t)win * 49 + i) * C_ + head * HD_ + t * 16 + nl] =
              (_Float16)acc[t][r];
      }
  }
}

// ---------------- workspace layout (bytes, all 256-aligned) ----------------
static const size_t O_XW   = 0;                               // f16 [NTOK, 256]
static const size_t O_QKV  = (size_t)NTOK_ * 256 * 2;         // f16 [NTOK, 768]
static const size_t O_ATTN = O_QKV + (size_t)NTOK_ * 768 * 2; // f16 [NTOK, 256]
static const size_t O_X2   = O_ATTN + (size_t)NTOK_ * 256 * 2;// f32 [NTOK, 256]
static const size_t O_H16  = O_X2 + (size_t)NTOK_ * 256 * 4;  // f16 [NTOK, 256]
static const size_t O_WQKV = O_H16 + (size_t)NTOK_ * 256 * 2; // packed f16 weights
static const size_t O_WPROJ= O_WQKV + (size_t)C_ * 3 * C_ * 2;
static const size_t O_W1   = O_WPROJ + (size_t)C_ * C_ * 2;
static const size_t O_W2   = O_W1 + (size_t)C_ * HID_ * 2;
static const size_t O_G16  = 0;  // f16 [NTOK,1024]: aliases retired XW+QKV (exact fit)

extern "C" void kernel_launch(void* const* d_in, const int* in_sizes, int n_in,
                              void* d_out, int out_size, void* d_ws, size_t ws_size,
                              hipStream_t stream) {
  const float* x        = (const float*)d_in[0];
  const float* ln1_g    = (const float*)d_in[1];
  const float* ln1_b    = (const float*)d_in[2];
  const float* qkv_w    = (const float*)d_in[3];
  const float* qkv_b    = (const float*)d_in[4];
  const float* proj_w   = (const float*)d_in[5];
  const float* proj_b   = (const float*)d_in[6];
  const float* bias_tab = (const float*)d_in[7];
  const float* ln2_g    = (const float*)d_in[8];
  const float* ln2_b    = (const float*)d_in[9];
  const float* mlp_w1   = (const float*)d_in[10];
  const float* mlp_b1   = (const float*)d_in[11];
  const float* mlp_w2   = (const float*)d_in[12];
  const float* mlp_b2   = (const float*)d_in[13];

  char* ws = (char*)d_ws;
  _Float16* xw16   = (_Float16*)(ws + O_XW);
  _Float16* qkv16  = (_Float16*)(ws + O_QKV);
  _Float16* attn16 = (_Float16*)(ws + O_ATTN);
  float*    x2     = (float*)   (ws + O_X2);
  _Float16* h16    = (_Float16*)(ws + O_H16);
  _Float16* g16    = (_Float16*)(ws + O_G16);
  _Float16* wqkv16 = (_Float16*)(ws + O_WQKV);
  _Float16* wproj16= (_Float16*)(ws + O_WPROJ);
  _Float16* w116   = (_Float16*)(ws + O_W1);
  _Float16* w216   = (_Float16*)(ws + O_W2);

  // cast + repack weights into WMMA B-fragment tile order (tiny, L2 resident)
  pack_w16<<<(C_ * 3 * C_ / 16 + 255) / 256, 256, 0, stream>>>(qkv_w, wqkv16, C_, 3 * C_);
  pack_w16<<<(C_ * C_   / 16 + 255) / 256, 256, 0, stream>>>(proj_w, wproj16, C_, C_);
  pack_w16<<<(C_ * HID_ / 16 + 255) / 256, 256, 0, stream>>>(mlp_w1, w116, C_, HID_);
  pack_w16<<<(HID_ * C_ / 16 + 255) / 256, 256, 0, stream>>>(mlp_w2, w216, HID_, C_);

  // LN1 + roll(-3,-3) + window partition -> f16 [win-token, C]
  lnorm_kernel<true><<<NTOK_, 256, 0, stream>>>(x, ln1_g, ln1_b, xw16);

  // qkv: [100352,256] @ [256,768]
  gemm_bias_f16<false><<<dim3(768 / 64, NTOK_ / 256), 256, 0, stream>>>(
      xw16, wqkv16, qkv_b, qkv16, 768, C_);

  // windowed attention, one block per (window, head)
  attn_kernel<<<dim3(NWIN_, NH_), 128, 0, stream>>>(qkv16, bias_tab, attn16);

  // proj + window-reverse + roll(+3,+3) + residual -> x2 (fp32)
  gemm_proj_scatter<<<dim3(C_ / 64, NTOK_ / 256), 256, 0, stream>>>(
      attn16, wproj16, proj_b, x, x2, C_, C_);

  // LN2 -> f16
  lnorm_kernel<false><<<NTOK_, 256, 0, stream>>>(x2, ln2_g, ln2_b, h16);

  // MLP1 + exact GELU: [100352,256] @ [256,1024]
  gemm_bias_f16<true><<<dim3(HID_ / 64, NTOK_ / 256), 256, 0, stream>>>(
      h16, w116, mlp_b1, g16, HID_, C_);

  // MLP2 + residual: [100352,1024] @ [1024,256] + x2 -> out fp32
  gemm_mlp2_out<<<dim3(C_ / 64, NTOK_ / 256), 256, 0, stream>>>(
      g16, w216, mlp_b2, x2, (float*)d_out, C_, HID_);
}
